// qMoEModelBatched_39479339385168
// MI455X (gfx1250) — compile-verified, matrix operand order checked by reading, not run on hardware
//
#include <hip/hip_runtime.h>
#include <hip/hip_bf16.h>
#include <math.h>

// ---------------------------------------------------------------------------
// MoE forward for MI455X (gfx1250, wave32, WMMA).
// Compute-bound (~0.9 TFLOP dense-equivalent): convert to bf16 and drive
// v_wmma_f32_16x16x32_bf16. Weights (~105MB bf16) fit in the 192MB L2.
// GEMM staging: gfx1250 async global->LDS DMA (ASYNCcnt), 3-buffer ring with
// 2-deep prefetch so each tile's DMA has two full compute blocks to complete.
// ---------------------------------------------------------------------------

#if defined(__has_builtin)
#if __has_builtin(__builtin_amdgcn_global_load_async_to_lds_b128) && \
    __has_builtin(__builtin_amdgcn_s_wait_asynccnt)
#define USE_ASYNC_LDS 1
#endif
#endif

typedef int v4i __attribute__((ext_vector_type(4)));
// builtin signature (from hipcc diagnostic): (int4 __device__*, int4 __shared__*, imm, imm)
#define GLOBAL_AS4(p) ((__attribute__((address_space(1))) v4i*)(p))
#define LDS_AS4(p)    ((__attribute__((address_space(3))) v4i*)(p))

namespace {
constexpr int B_  = 8192;
constexpr int D_  = 1024;
constexpr int C_  = 527;
constexpr int E_  = 4;
constexpr int HR_ = 512;
constexpr int H1_ = 4096;
constexpr int H2_ = 2048;
constexpr int CP_ = 640;     // C padded up to a multiple of NBLK (128*5)

constexpr int MBLK  = 128;
constexpr int NBLK  = 128;
constexpr int KSTEP = 32;
constexpr int LDSS  = KSTEP + 8;   // 40 bf16 row stride -> 80B: 16 lanes' b128
                                   // fragment loads cover all 64 LDS banks once
constexpr int NBUF  = 3;           // 3-stage async pipeline (61,440 B LDS)
}

typedef __bf16 v4bf  __attribute__((ext_vector_type(4)));
typedef __bf16 v8bf  __attribute__((ext_vector_type(8)));
typedef __bf16 v16bf __attribute__((ext_vector_type(16)));
typedef float  v8f   __attribute__((ext_vector_type(8)));

union Frag {                 // 16 bf16 per lane = two 16B LDS b128 loads
  v16bf v;
  struct { v8bf lo, hi; } h;
};

// ---------------------------------------------------------------------------
// f32 -> bf16 straight conversion (for activations x)
// ---------------------------------------------------------------------------
__global__ __launch_bounds__(256) void cvt_f32_bf16(
    const float* __restrict__ s, __bf16* __restrict__ d, long n4) {
  long i = (long)blockIdx.x * blockDim.x + threadIdx.x;
  const long stride = (long)gridDim.x * blockDim.x;
  for (; i < n4; i += stride) {
    const float4 v = ((const float4*)s)[i];
    v4bf o;
    o[0] = (__bf16)v.x; o[1] = (__bf16)v.y; o[2] = (__bf16)v.z; o[3] = (__bf16)v.w;
    ((v4bf*)d)[i] = o;
  }
}

// ---------------------------------------------------------------------------
// f32 [K,N] -> bf16 [Npad,K] transpose+convert (weights become B^T so GEMM
// B-fragments are K-contiguous; rows n>=N zero-filled for padded tails).
// ---------------------------------------------------------------------------
__global__ __launch_bounds__(256) void cvtT_f32_bf16(
    const float* __restrict__ src, __bf16* __restrict__ dst,
    int K, int N, int Npad, long sBatch, long dBatch) {
  __shared__ __bf16 tile[32][33];
  src += (long)blockIdx.z * sBatch;
  dst += (long)blockIdx.z * dBatch;
  const int k0 = blockIdx.x * 32, n0 = blockIdx.y * 32;
  const int tx = threadIdx.x & 31, ty = threadIdx.x >> 5;
  #pragma unroll
  for (int r = ty; r < 32; r += 8) {
    const int n = n0 + tx;
    const float v = (n < N) ? src[(long)(k0 + r) * N + n] : 0.f;
    tile[r][tx] = (__bf16)v;
  }
  __syncthreads();
  #pragma unroll
  for (int r = ty; r < 32; r += 8) {
    const int n = n0 + r;
    if (n < Npad) dst[(long)n * K + (k0 + tx)] = tile[tx][r];
  }
}

// ---------------------------------------------------------------------------
// Tiled bf16 WMMA GEMM:  out = epilogue(A[M,K] @ Bt[Np,K]^T + bias)
//   MODE 0: out bf16 [M,Np]   = relu(acc + bias)              (hidden layers)
//   MODE 1: out f32  [M,NOUT] (+)= gates[m,e]*0.5*(acc+bias)  (final layer)
// Workgroup: 256 threads = 8 wave32 in 2(M)x4(N); wave tile 64x32 = 4x2 WMMA.
// LDS: 3-buffer ring of 128x32 bf16 tiles for A and Bt (60KB), filled by
// global_load_async_to_lds_b128 (4 async ops/wave per tile, 2 tiles in flight).
// ---------------------------------------------------------------------------
template <int MODE>
__global__ __launch_bounds__(256) void gemm_bf16(
    const __bf16* __restrict__ A,
    const __bf16* __restrict__ Bt,
    const float*  __restrict__ bias,
    void* __restrict__ outp,
    const float* __restrict__ gates,
    int K, int Np, int NOUT, int expert, int accum) {
  __shared__ __bf16 lA[NBUF][MBLK * LDSS];
  __shared__ __bf16 lB[NBUF][NBLK * LDSS];

  const int tid  = threadIdx.x;
  const int lane = tid & 31;
  const int wave = tid >> 5;
  const int wm   = wave >> 2;      // 0..1
  const int wn   = wave & 3;       // 0..3
  const int lr   = lane & 15;
  const int half = lane >> 4;
  const int m0   = blockIdx.y * MBLK;
  const int n0   = blockIdx.x * NBLK;

  // staging: 128 rows x 32 bf16 = 4 x 16B chunks/row; 256 threads, 2 iters
  const int srow = tid >> 2;          // 0..63
  const int scol = (tid & 3) * 8;     // bf16 units

  v8f acc[4][2];
  #pragma unroll
  for (int mt = 0; mt < 4; ++mt)
    #pragma unroll
    for (int nt = 0; nt < 2; ++nt)
      #pragma unroll
      for (int i = 0; i < 8; ++i) acc[mt][nt][i] = 0.f;

#if USE_ASYNC_LDS
  // DMA tiles directly into LDS: no data VGPRs, no ds_store, no spills.
  auto issueTile = [&](int kc, int buf) {
    const int k0 = kc * KSTEP;
    #pragma unroll
    for (int it = 0; it < 2; ++it) {
      const int r = srow + it * 64;
      __builtin_amdgcn_global_load_async_to_lds_b128(
          GLOBAL_AS4(A + (long)(m0 + r) * K + k0 + scol),
          LDS_AS4(&lA[buf][r * LDSS + scol]), 0, 0);
      __builtin_amdgcn_global_load_async_to_lds_b128(
          GLOBAL_AS4(Bt + (long)(n0 + r) * K + k0 + scol),
          LDS_AS4(&lB[buf][r * LDSS + scol]), 0, 0);
    }
  };
#else
  float4 ra[2], rb[2];
  auto loadTile = [&](int kc) {
    const int k0 = kc * KSTEP;
    #pragma unroll
    for (int it = 0; it < 2; ++it) {
      const int r = srow + it * 64;
      ra[it] = *(const float4*)(A  + (long)(m0 + r) * K + k0 + scol);
      rb[it] = *(const float4*)(Bt + (long)(n0 + r) * K + k0 + scol);
    }
  };
  auto storeTile = [&](int buf) {
    #pragma unroll
    for (int it = 0; it < 2; ++it) {
      const int r = srow + it * 64;
      *(float4*)&lA[buf][r * LDSS + scol] = ra[it];
      *(float4*)&lB[buf][r * LDSS + scol] = rb[it];
    }
  };
#endif

  const int nk = K / KSTEP;   // >= 32 for all layers here

#if USE_ASYNC_LDS
  issueTile(0, 0);
  issueTile(1, 1);
  int cur = 0, ibuf = 2;      // ibuf = buffer for tile kc+2
#else
  loadTile(0);
  storeTile(0);
  __syncthreads();
#endif

  for (int kc = 0; kc < nk; ++kc) {
#if USE_ASYNC_LDS
    if (kc + 2 < nk) issueTile(kc + 2, ibuf);
    // async ops retire in order per wave: tiles newer than `cur` hold
    // 4 ops each, so waiting to <= 4*inflight drains the current tile.
    if (kc + 2 < nk)      __builtin_amdgcn_s_wait_asynccnt(8);
    else if (kc + 1 < nk) __builtin_amdgcn_s_wait_asynccnt(4);
    else                  __builtin_amdgcn_s_wait_asynccnt(0);
    __syncthreads();             // all waves' DMA for buf `cur` visible
#else
    const int cur = kc & 1;
    if (kc + 1 < nk) loadTile(kc + 1);   // prefetch next tile into VGPRs
#endif

    // per-lane fragment layout (16-bit A 16x32): VGPR v<4 hold k=half*8+0..7,
    // v>=4 hold k=16+half*8+0..7  ->  two b128 loads at kb and kb+16
    Frag a[4], b[2];
    const int kb = half * 8;
    #pragma unroll
    for (int mt = 0; mt < 4; ++mt) {
      const __bf16* p = &lA[cur][(wm * 64 + mt * 16 + lr) * LDSS + kb];
      a[mt].h.lo = *(const v8bf*)p;
      a[mt].h.hi = *(const v8bf*)(p + 16);
    }
    #pragma unroll
    for (int nt = 0; nt < 2; ++nt) {
      const __bf16* p = &lB[cur][(wn * 32 + nt * 16 + lr) * LDSS + kb];
      b[nt].h.lo = *(const v8bf*)p;
      b[nt].h.hi = *(const v8bf*)(p + 16);
    }
    #pragma unroll
    for (int mt = 0; mt < 4; ++mt)
      #pragma unroll
      for (int nt = 0; nt < 2; ++nt)
        acc[mt][nt] = __builtin_amdgcn_wmma_f32_16x16x32_bf16(
            false, a[mt].v, false, b[nt].v, (short)0, acc[mt][nt], false, false);

    __syncthreads();             // done reading `cur` before it is refilled
#if USE_ASYNC_LDS
    cur  = (cur  == NBUF - 1) ? 0 : cur + 1;
    ibuf = (ibuf == NBUF - 1) ? 0 : ibuf + 1;
#else
    if (kc + 1 < nk) storeTile(cur ^ 1);
    __syncthreads();
#endif
  }

  // epilogue; C/D layout: VGPR i -> M = i + 8*half, N = lane%16
  #pragma unroll
  for (int mt = 0; mt < 4; ++mt) {
    #pragma unroll
    for (int nt = 0; nt < 2; ++nt) {
      const int gm = m0 + wm * 64 + mt * 16 + half * 8;
      const int gn = n0 + wn * 32 + nt * 16 + lr;
      if (MODE == 0) {
        __bf16* out = (__bf16*)outp;
        const float bv = bias[gn];
        #pragma unroll
        for (int i = 0; i < 8; ++i) {
          float v = acc[mt][nt][i] + bv;
          v = v > 0.f ? v : 0.f;
          out[(long)(gm + i) * Np + gn] = (__bf16)v;
        }
      } else {
        float* out = (float*)outp;
        const bool ok = gn < NOUT;
        const float bv = ok ? bias[gn] : 0.f;
        #pragma unroll
        for (int i = 0; i < 8; ++i) {
          const int m = gm + i;
          const float g = gates[m * E_ + expert] * 0.5f;   // / top_k
          const float v = (acc[mt][nt][i] + bv) * g;
          if (ok) {
            const long idx = (long)m * NOUT + gn;
            out[idx] = accum ? (out[idx] + v) : v;
          }
        }
      }
    }
  }
}

// ---------------------------------------------------------------------------
// Router tail: logits = Hr[b,:] @ Wr2 + br2 (HR=512 x E=4), softmax, top-2.
// One wave per row. Also writes router_p and lb_loss=0.
// ---------------------------------------------------------------------------
__global__ __launch_bounds__(256) void router_kernel(
    const __bf16* __restrict__ Hr, const float* __restrict__ Wr2,
    const float* __restrict__ br2, float* __restrict__ router_p,
    float* __restrict__ gates, float* __restrict__ lb) {
  const int lane = threadIdx.x & 31;
  const int wave = threadIdx.x >> 5;
  const int b = blockIdx.x * 8 + wave;
  float a0 = 0.f, a1 = 0.f, a2 = 0.f, a3 = 0.f;
  for (int h = lane; h < HR_; h += 32) {
    const float hv = (float)Hr[(long)b * HR_ + h];
    const float4 w = *(const float4*)(Wr2 + h * 4);
    a0 += hv * w.x; a1 += hv * w.y; a2 += hv * w.z; a3 += hv * w.w;
  }
  #pragma unroll
  for (int off = 16; off > 0; off >>= 1) {
    a0 += __shfl_down(a0, off, 32);
    a1 += __shfl_down(a1, off, 32);
    a2 += __shfl_down(a2, off, 32);
    a3 += __shfl_down(a3, off, 32);
  }
  if (lane == 0) {
    float lg[4] = { a0 + br2[0], a1 + br2[1], a2 + br2[2], a3 + br2[3] };
    const float mx = fmaxf(fmaxf(lg[0], lg[1]), fmaxf(lg[2], lg[3]));
    float p[4], s = 0.f;
    #pragma unroll
    for (int e = 0; e < 4; ++e) { p[e] = __expf(lg[e] - mx); s += p[e]; }
    const float inv = 1.f / s;
    #pragma unroll
    for (int e = 0; e < 4; ++e) { p[e] *= inv; router_p[(long)b * 4 + e] = p[e]; }
    int i1 = 0;
    for (int e = 1; e < 4; ++e) if (p[e] > p[i1]) i1 = e;     // first max
    int i2 = -1;
    for (int e = 0; e < 4; ++e) if (e != i1 && (i2 < 0 || p[e] > p[i2])) i2 = e;
    #pragma unroll
    for (int e = 0; e < 4; ++e)
      gates[(long)b * 4 + e] = (e == i1 || e == i2) ? p[e] : 0.f;
  }
  if (blockIdx.x == 0 && threadIdx.x == 0) *lb = 0.f;
}

// ---------------------------------------------------------------------------
extern "C" void kernel_launch(void* const* d_in, const int* in_sizes, int n_in,
                              void* d_out, int out_size, void* d_ws, size_t ws_size,
                              hipStream_t stream) {
  (void)in_sizes; (void)n_in; (void)out_size; (void)ws_size;

  const float* x   = (const float*)d_in[0];
  const float* Wr1 = (const float*)d_in[1];
  const float* br1 = (const float*)d_in[2];
  const float* Wr2 = (const float*)d_in[3];
  const float* br2 = (const float*)d_in[4];
  const float* We1 = (const float*)d_in[5];
  const float* be1 = (const float*)d_in[6];
  const float* We2 = (const float*)d_in[7];
  const float* be2 = (const float*)d_in[8];
  const float* We3 = (const float*)d_in[9];
  const float* be3 = (const float*)d_in[10];
  // d_in[11] = top_k (hardcoded 2)

  float* out      = (float*)d_out;                    // [B, C]
  float* router_p = out + (size_t)B_ * C_;            // [B, E]
  float* lb       = router_p + (size_t)B_ * E_;       // scalar

  // workspace carve-out (~240 MB)
  char* ws = (char*)d_ws;
  size_t off = 0;
  auto carve = [&](size_t bytes) {
    char* p = ws + off;
    off += (bytes + 255) & ~(size_t)255;
    return p;
  };
  __bf16* xb    = (__bf16*)carve((size_t)B_ * D_ * 2);
  __bf16* Wr1t  = (__bf16*)carve((size_t)HR_ * D_ * 2);
  __bf16* We1t  = (__bf16*)carve((size_t)E_ * H1_ * D_ * 2);
  __bf16* We2t  = (__bf16*)carve((size_t)E_ * H2_ * H1_ * 2);
  __bf16* We3t  = (__bf16*)carve((size_t)E_ * CP_ * H2_ * 2);
  __bf16* Hrb   = (__bf16*)carve((size_t)B_ * HR_ * 2);
  __bf16* H1b   = (__bf16*)carve((size_t)B_ * H1_ * 2);
  __bf16* H2b   = (__bf16*)carve((size_t)B_ * H2_ * 2);
  float*  gates = (float*) carve((size_t)B_ * E_ * 4);

  const dim3 blk(256);

  // bf16 conversions (x straight; weights transposed to [N,K])
  cvt_f32_bf16<<<2048, blk, 0, stream>>>(x, xb, (long)B_ * D_ / 4);
  cvtT_f32_bf16<<<dim3(D_ / 32, HR_ / 32, 1), blk, 0, stream>>>(
      Wr1, Wr1t, D_, HR_, HR_, 0, 0);
  cvtT_f32_bf16<<<dim3(D_ / 32, H1_ / 32, E_), blk, 0, stream>>>(
      We1, We1t, D_, H1_, H1_, (long)D_ * H1_, (long)H1_ * D_);
  cvtT_f32_bf16<<<dim3(H1_ / 32, H2_ / 32, E_), blk, 0, stream>>>(
      We2, We2t, H1_, H2_, H2_, (long)H1_ * H2_, (long)H2_ * H1_);
  cvtT_f32_bf16<<<dim3(H2_ / 32, CP_ / 32, E_), blk, 0, stream>>>(
      We3, We3t, H2_, C_, CP_, (long)H2_ * C_, (long)CP_ * H2_);

  // router hidden: Hr = relu(x @ Wr1 + br1)
  gemm_bf16<0><<<dim3(HR_ / NBLK, B_ / MBLK), blk, 0, stream>>>(
      xb, Wr1t, br1, Hrb, nullptr, D_, HR_, HR_, 0, 0);
  // router_p, gates, lb_loss
  router_kernel<<<B_ / 8, blk, 0, stream>>>(Hrb, Wr2, br2, router_p, gates, lb);

  // experts (sequential -> H1b/H2b reused; expert 0 writes out, rest accumulate)
  for (int e = 0; e < E_; ++e) {
    gemm_bf16<0><<<dim3(H1_ / NBLK, B_ / MBLK), blk, 0, stream>>>(
        xb, We1t + (size_t)e * H1_ * D_, be1 + (size_t)e * H1_,
        H1b, nullptr, D_, H1_, H1_, 0, 0);
    gemm_bf16<0><<<dim3(H2_ / NBLK, B_ / MBLK), blk, 0, stream>>>(
        H1b, We2t + (size_t)e * H2_ * H1_, be2 + (size_t)e * H2_,
        H2b, nullptr, H1_, H2_, H2_, 0, 0);
    gemm_bf16<1><<<dim3(CP_ / NBLK, B_ / MBLK), blk, 0, stream>>>(
        H2b, We3t + (size_t)e * CP_ * H2_, be3 + (size_t)e * C_,
        out, gates, H2_, CP_, C_, e, e > 0);
  }
}
